// SelfAttentionLayer_16037407883871
// MI455X (gfx1250) — matrix operand order repaired and checked by reading, not compiled, not run on hardware
//
#include <hip/hip_runtime.h>
#include <hip/hip_bf16.h>

// ---------------------------------------------------------------------------
// SelfAttentionLayer forward for MI455X (gfx1250, wave32, WMMA).
//
// Shapes (fixed by setup_inputs): B=128, NINS=4096, NHID=128, NH2=64.
// Math trick: concat([own, inputs]) @ W1 == inputs @ W1[128:,:] + own @ W1[:128,:]
// where "own @ W1_top" is a per-batch 64-vector -> folded into the bias.
// Main GEMM: [B*4096, 128](f16) x [128, 64](f16) -> f32, via
// v_wmma_f32_16x16x32_f16 (4 k-steps x 4 n-tiles per 16-row tile).
// ---------------------------------------------------------------------------

typedef _Float16 v16h __attribute__((ext_vector_type(16)));
typedef float    v8f  __attribute__((ext_vector_type(8)));

union AFrag { v16h v; _Float16 h[16]; };
union CFrag { v8f  v; float    f[8];  };

#define B_DIM    128
#define NINS     4096
#define NHID     128
#define NH2      64
#define ROWS_PER_BLOCK 512   // 32 tiles of 16 rows, 8 waves -> 4 tiles/wave
#define NCHUNKS  (NINS / ROWS_PER_BLOCK)   // 8

// ---------------------------------------------------------------------------
// Kernel 1: attention scores att[b, n] via WMMA GEMM + fused epilogue.
// ---------------------------------------------------------------------------
__global__ __launch_bounds__(256)
void attn_score_kernel(const float* __restrict__ inputs,   // [B, NINS, NHID]
                       const int*   __restrict__ pIndex,   // scalar
                       const float* __restrict__ W1,       // [2*NHID, NH2]
                       const float* __restrict__ b1,       // [NH2]
                       const float* __restrict__ W2,       // [NH2]
                       const float* __restrict__ b2,       // [1]
                       float*       __restrict__ att)      // [B, NINS]
{
    // LDS: W1 bottom half, transposed to [k][d] as f16 so each lane's
    // B-fragment slice (16 consecutive d) is one contiguous 32B LDS read.
    __shared__ __attribute__((aligned(32))) _Float16 w1t[NH2 * NHID]; // 16 KB
    __shared__ float s_bias[NH2];  // ownProj + b1
    __shared__ float s_w2[NH2];

    const int tid   = threadIdx.x;
    const int batch = blockIdx.y;
    const int nbase = blockIdx.x * ROWS_PER_BLOCK;
    const int idx   = *pIndex;

    // Stage W1[128+d][k] -> w1t[k*NHID + d] as f16 (coalesced global reads).
    for (int e = tid; e < NH2 * NHID; e += 256) {
        const int k = e & (NH2 - 1);
        const int d = e >> 6;
        w1t[k * NHID + d] = (_Float16)W1[(NHID + d) * NH2 + k]; // = W1[2*NHID*? ] row-major
    }

    // Per-batch own-projection folded into bias (fp32, exact, cheap: 128x64 MACs).
    if (tid < NH2) {
        const float* own = inputs + ((size_t)batch * NINS + idx) * NHID;
        float acc = 0.0f;
        #pragma unroll 4
        for (int d = 0; d < NHID; ++d)
            acc = fmaf(own[d], W1[d * NH2 + tid], acc);
        s_bias[tid] = acc + b1[tid];
        s_w2[tid]   = W2[tid];
    }
    __syncthreads();

    const int lane = tid & 31;
    const int wave = tid >> 5;
    const int col  = lane & 15;   // N column within tile / M row for A
    const int hi   = lane >> 4;   // half-wave selector

    // Preload all B-fragments (k-independent of the row tile).
    // f16 B 32x16 as loaded here (contiguous K, same order as A): lanes 0-15 hold K=0..15,
    // lanes 16-31 hold K=16..31, column N = lane%16, 2 halves per VGPR.
    AFrag bf[16];
    #pragma unroll
    for (int kk = 0; kk < 4; ++kk) {
        #pragma unroll
        for (int nt = 0; nt < 4; ++nt) {
            const int kcol = nt * 16 + col;
            const int doff = kk * 32 + hi * 16;
            bf[kk * 4 + nt].v =
                *reinterpret_cast<const v16h*>(&w1t[kcol * NHID + doff]);
        }
    }

    const float b2v = b2[0];

    for (int t = wave; t < ROWS_PER_BLOCK / 16; t += 8) {
        const int n0 = nbase + t * 16;
        const float* rowp =
            inputs + ((size_t)batch * NINS + n0 + col) * NHID;

        CFrag acc[4] = {};

        #pragma unroll
        for (int kk = 0; kk < 4; ++kk) {
            // f16 A 16x32 layout: lanes 0-15 (M=0..15): K = kbase+{0..15};
            // lanes 16-31 (M=0..15): K = kbase+{16..31}.
            const float* p0 = rowp + kk * 32 + hi * 16;
            const float4 f0 = *reinterpret_cast<const float4*>(p0);
            const float4 f1 = *reinterpret_cast<const float4*>(p0 + 4);
            const float4 f2 = *reinterpret_cast<const float4*>(p0 + 8);
            const float4 f3 = *reinterpret_cast<const float4*>(p0 + 12);

            AFrag a;
            a.h[0]  = (_Float16)f0.x; a.h[1]  = (_Float16)f0.y;
            a.h[2]  = (_Float16)f0.z; a.h[3]  = (_Float16)f0.w;
            a.h[4]  = (_Float16)f1.x; a.h[5]  = (_Float16)f1.y;
            a.h[6]  = (_Float16)f1.z; a.h[7]  = (_Float16)f1.w;
            a.h[8]  = (_Float16)f2.x; a.h[9]  = (_Float16)f2.y;
            a.h[10] = (_Float16)f2.z; a.h[11] = (_Float16)f2.w;
            a.h[12] = (_Float16)f3.x; a.h[13] = (_Float16)f3.y;
            a.h[14] = (_Float16)f3.z; a.h[15] = (_Float16)f3.w;

            #pragma unroll
            for (int nt = 0; nt < 4; ++nt) {
                acc[nt].v = __builtin_amdgcn_wmma_f32_16x16x32_f16(
                    /*neg_a=*/false, a.v,
                    /*neg_b=*/false, bf[kk * 4 + nt].v,
                    /*c_mod=*/(short)0, acc[nt].v,
                    /*reuse_a=*/false, /*reuse_b=*/false);
            }
        }

        // Fused epilogue: h = relu(c + ownProj + b1); att_partial = h . W2.
        // C/D layout: VGPR r, lanes 0-15 -> (M=r, N=lane); lanes 16-31 -> (M=r+8).
        float part[8] = {0,0,0,0,0,0,0,0};
        #pragma unroll
        for (int nt = 0; nt < 4; ++nt) {
            const int k    = nt * 16 + col;
            const float bs = s_bias[k];
            const float wv = s_w2[k];
            #pragma unroll
            for (int r = 0; r < 8; ++r) {
                float hv = fmaxf(acc[nt].f[r] + bs, 0.0f);
                part[r] = fmaf(hv, wv, part[r]);
            }
        }
        // Reduce the remaining 16 N-columns across each half-wave.
        #pragma unroll
        for (int m = 1; m < 16; m <<= 1) {
            #pragma unroll
            for (int r = 0; r < 8; ++r)
                part[r] += __shfl_xor(part[r], m, 32);
        }
        if (col == 0) {
            #pragma unroll
            for (int r = 0; r < 8; ++r)
                att[(size_t)batch * NINS + n0 + hi * 8 + r] = part[r] + b2v;
        }
    }
}

// ---------------------------------------------------------------------------
// Kernel 2: per-batch softmax over NINS + weighted sum over inputs.
// One block per batch; per-batch input slice (2 MiB) is L2-resident.
// ---------------------------------------------------------------------------
__global__ __launch_bounds__(256)
void attn_softmax_out_kernel(const float* __restrict__ inputs, // [B, NINS, NHID]
                             const float* __restrict__ att,    // [B, NINS]
                             float*       __restrict__ out)    // [B, NHID]
{
    __shared__ float sw[NINS];    // 16 KB softmax weights
    __shared__ float red[256];

    const int b = blockIdx.x;
    const int t = threadIdx.x;
    const float* attb = att + (size_t)b * NINS;

    // 1) max
    float m = -3.402823466e38f;
    for (int n = t; n < NINS; n += 256) m = fmaxf(m, attb[n]);
    red[t] = m; __syncthreads();
    for (int s = 128; s > 0; s >>= 1) {
        if (t < s) red[t] = fmaxf(red[t], red[t + s]);
        __syncthreads();
    }
    const float gmax = red[0];
    __syncthreads();

    // 2) exp + sum
    float sum = 0.0f;
    for (int n = t; n < NINS; n += 256) {
        const float e = __expf(attb[n] - gmax);
        sw[n] = e;
        sum += e;
    }
    red[t] = sum; __syncthreads();
    for (int s = 128; s > 0; s >>= 1) {
        if (t < s) red[t] += red[t + s];
        __syncthreads();
    }
    const float inv = 1.0f / red[0];
    __syncthreads();

    // 3) out[b,d] = inv * sum_n sw[n] * inputs[b,n,d]
    const int d    = t & (NHID - 1);
    const int part = t >> 7;                 // 0 or 1
    const float* inb = inputs + (size_t)b * NINS * NHID;
    float acc = 0.0f;
    for (int n = part; n < NINS; n += 2)
        acc = fmaf(sw[n], inb[(size_t)n * NHID + d], acc);
    red[t] = acc; __syncthreads();
    if (part == 0)
        out[(size_t)b * NHID + d] = (red[t] + red[t + NHID]) * inv;
}

// ---------------------------------------------------------------------------
extern "C" void kernel_launch(void* const* d_in, const int* in_sizes, int n_in,
                              void* d_out, int out_size, void* d_ws, size_t ws_size,
                              hipStream_t stream) {
    const float* inputs = (const float*)d_in[0];
    const int*   pIndex = (const int*)  d_in[1];
    // d_in[2] (claims) is unused by the forward pass.
    const float* W1     = (const float*)d_in[3];
    const float* b1     = (const float*)d_in[4];
    const float* W2     = (const float*)d_in[5];
    const float* b2     = (const float*)d_in[6];
    float*       out    = (float*)d_out;
    float*       att    = (float*)d_ws;   // B*NINS floats = 2 MiB scratch

    dim3 grid1(NCHUNKS, B_DIM);           // 8 x 128 blocks, 256 threads (8 waves)
    attn_score_kernel<<<grid1, 256, 0, stream>>>(inputs, pIndex, W1, b1, W2, b2, att);
    attn_softmax_out_kernel<<<B_DIM, 256, 0, stream>>>(inputs, att, out);
}
